// MLPConv2d_6682969113002
// MI455X (gfx1250) — matrix operand order repaired and checked by reading, not compiled, not run on hardware
//
#include <hip/hip_runtime.h>
#include <cstdint>
#include <cstddef>

typedef __attribute__((ext_vector_type(16))) _Float16 v16h;
typedef __attribute__((ext_vector_type(2)))  __fp16   v2fp16;
typedef __attribute__((ext_vector_type(8)))  float    v8f;

// ---- problem geometry -------------------------------------------------------
#define C_IN   3
#define H_IN   224
#define W_IN   224
#define BATCH  32
#define H_OUT  109
#define W_OUT  109
#define LPOS   (H_OUT * W_OUT)      // 11881 positions / image
#define M_TOT  (BATCH * LPOS)       // 380192 rows (divisible by 16)
#define D1     147                  // 3*7*7
#define N1     200
#define N2     50
#define N3     64

// ---- padded WMMA dims (16x16x32 f16 tiles) ---------------------------------
#define NT1 13   // N1 -> 208
#define KC1 5    // D1 -> 160
#define NT2 4    // N2 -> 64
#define KC2 7    // 200 -> 224
#define NT3 4    // 64
#define KC3 2    // 50 -> 64

// ---- LDS layout (bytes); all offsets 16B aligned ---------------------------
#define W1S_OFF 0
#define W1S_CNT (NT1 * KC1 * 32 * 8)               // 16640 u32 = 66560 B
#define W2S_OFF (W1S_OFF + W1S_CNT * 4)
#define W2S_CNT (NT2 * KC2 * 32 * 8)               // 7168 u32 = 28672 B
#define W3S_OFF (W2S_OFF + W2S_CNT * 4)
#define W3S_CNT (NT3 * KC3 * 32 * 8)               // 2048 u32 = 8192 B
#define B1S_OFF (W3S_OFF + W3S_CNT * 4)
#define B2S_OFF (B1S_OFF + 208 * 4)
#define B3S_OFF (B2S_OFF + 64 * 4)
#define H1_OFF  (B3S_OFF + 64 * 4)
#define H1_STRIDE 224                               // halfs per row (448 B)
#define H2_OFF  (H1_OFF + 8 * 16 * H1_STRIDE * 2)
#define H2_STRIDE 64
#define SMEM_BYTES (H2_OFF + 8 * 16 * H2_STRIDE * 2)  // 178496 B (< 320 KB/WGP)

#define LDS_FENCE() asm volatile("s_wait_dscnt 0" ::: "memory")

union FragU { v16h h; uint32_t u[8]; uint4 q[2]; };
union AccU  { v8f  f; float    x[8]; };

// single-instruction f32x2 -> packed f16x2 (v_cvt_pk_rtz_f16_f32)
__device__ __forceinline__ uint32_t packf16(float a, float b) {
    v2fp16 p = __builtin_amdgcn_cvt_pkrtz(a, b);
    return __builtin_bit_cast(uint32_t, p);
}

// feature k of a patch anchored at xb (= &x[b][0][oy*2][ox*2]); k is not a
// compile-time constant: at the call site it also depends on hi = lane>>4.
__device__ __forceinline__ float load_feat(const float* xb, int k) {
    if (k >= D1) return 0.0f;
    int c  = k / 49;
    int rm = k - c * 49;
    int kh = rm / 7;
    int kw = rm - kh * 7;
    return xb[(size_t)c * (H_IN * W_IN) + kh * W_IN + kw];
}

__global__ __launch_bounds__(256) void mlpconv_fused_wmma(
    const float* __restrict__ x,
    const float* __restrict__ w1, const float* __restrict__ b1,
    const float* __restrict__ w2, const float* __restrict__ b2,
    const float* __restrict__ w3, const float* __restrict__ b3,
    float* __restrict__ out)
{
    extern __shared__ __align__(16) char smem[];
    uint32_t* w1s = (uint32_t*)(smem + W1S_OFF);
    uint32_t* w2s = (uint32_t*)(smem + W2S_OFF);
    uint32_t* w3s = (uint32_t*)(smem + W3S_OFF);
    float*    b1s = (float*)(smem + B1S_OFF);
    float*    b2s = (float*)(smem + B2S_OFF);
    float*    b3s = (float*)(smem + B3S_OFF);

    const int tid  = threadIdx.x;
    const int wave = tid >> 5;
    const int lane = tid & 31;
    const int hi   = lane >> 4;   // K-half selector (A and B layouts)
    const int ln   = lane & 15;   // M row (A) / N col (B,C)

    // ---- stage weights into LDS, f16, pre-swizzled into B-fragment order ---
    // fragment addr = ((nt*KC + kc)*32 + lane)*8 + v ; element pair
// (k, n),(k+1, n) with k = kc*32 + (v<4 ? (lane>>4)*8 + v*2 : 16 + (lane>>4)*8 + (v-4)*2), n = nt*16 + lane&15
    for (int idx = tid; idx < W1S_CNT; idx += 256) {
        int v = idx & 7, l = (idx >> 3) & 31, rest = idx >> 8;
        int kc = rest % KC1, nt = rest / KC1;
        int k = kc * 32 + ((v < 4) ? ((l >> 4) * 8 + v * 2) : (16 + (l >> 4) * 8 + (v - 4) * 2));
        int n = nt * 16 + (l & 15);
        float f0 = (k     < D1 && n < N1) ? w1[k * N1 + n]       : 0.0f;
        float f1 = (k + 1 < D1 && n < N1) ? w1[(k + 1) * N1 + n] : 0.0f;
        w1s[idx] = packf16(f0, f1);
    }
    for (int idx = tid; idx < W2S_CNT; idx += 256) {
        int v = idx & 7, l = (idx >> 3) & 31, rest = idx >> 8;
        int kc = rest % KC2, nt = rest / KC2;
        int k = kc * 32 + ((v < 4) ? ((l >> 4) * 8 + v * 2) : (16 + (l >> 4) * 8 + (v - 4) * 2));
        int n = nt * 16 + (l & 15);
        float f0 = (k     < N1 && n < N2) ? w2[k * N2 + n]       : 0.0f;
        float f1 = (k + 1 < N1 && n < N2) ? w2[(k + 1) * N2 + n] : 0.0f;
        w2s[idx] = packf16(f0, f1);
    }
    for (int idx = tid; idx < W3S_CNT; idx += 256) {
        int v = idx & 7, l = (idx >> 3) & 31, rest = idx >> 8;
        int kc = rest & 1, nt = rest >> 1;
        int k = kc * 32 + ((v < 4) ? ((l >> 4) * 8 + v * 2) : (16 + (l >> 4) * 8 + (v - 4) * 2));
        int n = nt * 16 + (l & 15);
        float f0 = (k     < N2) ? w3[k * N3 + n]       : 0.0f;
        float f1 = (k + 1 < N2) ? w3[(k + 1) * N3 + n] : 0.0f;
        w3s[idx] = packf16(f0, f1);
    }
    if (tid < 208) b1s[tid] = (tid < N1) ? b1[tid] : 0.0f;
    if (tid < 64)  b2s[tid] = (tid < N2) ? b2[tid] : 0.0f;
    if (tid < 64)  b3s[tid] = b3[tid];
    __syncthreads();

    _Float16* h1 = (_Float16*)(smem + H1_OFF) + (size_t)wave * 16 * H1_STRIDE;
    _Float16* h2 = (_Float16*)(smem + H2_OFF) + (size_t)wave * 16 * H2_STRIDE;

    const int numTiles = M_TOT / 16;   // 23762, exact
    for (int tile = blockIdx.x * 8 + wave; tile < numTiles;
         tile += gridDim.x * 8) {
        const int rowBase = tile * 16;

        // ---- fused im2col: build A1 fragments straight from global x ------
        int r  = rowBase + ln;                 // A layout: M = lane%16
        int rc = (r < M_TOT) ? r : (M_TOT - 1);
        int bb  = rc / LPOS;
        int pos = rc - bb * LPOS;
        int oy  = pos / W_OUT;
        int ox  = pos - oy * W_OUT;
        const float* xb = x + (size_t)bb * C_IN * H_IN * W_IN
                            + (size_t)(oy * 2) * W_IN + (ox * 2);
        uint32_t a1[KC1][8];
        #pragma unroll
        for (int kc = 0; kc < KC1; ++kc) {
            #pragma unroll
            for (int v = 0; v < 8; ++v) {
                // 16-bit A layout: v<4 -> K = hi*8 + 2v ; v>=4 -> +16
                int k = kc * 32 + ((v < 4) ? (hi * 8 + v * 2)
                                           : (16 + hi * 8 + (v - 4) * 2));
                a1[kc][v] = packf16(load_feat(xb, k), load_feat(xb, k + 1));
            }
        }

        // ---- layer 1: [16x160] x [160x208] + b1, relu -> h1 ---------------
        for (int nt = 0; nt < NT1; ++nt) {
            AccU acc;
            float bias = b1s[nt * 16 + ln];
            #pragma unroll
            for (int i = 0; i < 8; ++i) acc.x[i] = bias;
            #pragma unroll
            for (int kc = 0; kc < KC1; ++kc) {
                FragU a, bf;
                #pragma unroll
                for (int v = 0; v < 8; ++v) a.u[v] = a1[kc][v];
                const uint4* bp =
                    (const uint4*)(w1s + ((size_t)(nt * KC1 + kc) * 32 + lane) * 8);
                bf.q[0] = bp[0]; bf.q[1] = bp[1];
                acc.f = __builtin_amdgcn_wmma_f32_16x16x32_f16(
                    false, a.h, false, bf.h, (short)0, acc.f, false, false);
            }
            #pragma unroll
            for (int i = 0; i < 8; ++i) {       // C layout: M = hi*8 + i
                h1[(hi * 8 + i) * H1_STRIDE + nt * 16 + ln] =
                    (_Float16)fmaxf(acc.x[i], 0.0f);
            }
        }
        #pragma unroll
        for (int i = 0; i < 8; ++i)             // zero K-pad cols 208..223
            h1[(hi * 8 + i) * H1_STRIDE + 208 + ln] = (_Float16)0.0f;
        LDS_FENCE();

        // ---- layer 2: [16x224] x [224x64] + b2, relu -> h2 ----------------
        for (int nt = 0; nt < NT2; ++nt) {
            AccU acc;
            float bias = b2s[nt * 16 + ln];
            #pragma unroll
            for (int i = 0; i < 8; ++i) acc.x[i] = bias;
            #pragma unroll
            for (int kc = 0; kc < KC2; ++kc) {
                FragU a, bf;
                const uint4* ap =
                    (const uint4*)(h1 + ln * H1_STRIDE + kc * 32 + hi * 8);
                const uint4* ap2 =
                    (const uint4*)(h1 + ln * H1_STRIDE + kc * 32 + hi * 8 + 16);
                a.q[0] = ap[0]; a.q[1] = ap2[0];
                const uint4* bp =
                    (const uint4*)(w2s + ((size_t)(nt * KC2 + kc) * 32 + lane) * 8);
                bf.q[0] = bp[0]; bf.q[1] = bp[1];
                acc.f = __builtin_amdgcn_wmma_f32_16x16x32_f16(
                    false, a.h, false, bf.h, (short)0, acc.f, false, false);
            }
            #pragma unroll
            for (int i = 0; i < 8; ++i)
                h2[(hi * 8 + i) * H2_STRIDE + nt * 16 + ln] =
                    (_Float16)fmaxf(acc.x[i], 0.0f);
        }
        LDS_FENCE();

        // ---- layer 3: [16x64] x [64x64] + b3, sigmoid -> global out -------
        for (int nt = 0; nt < NT3; ++nt) {
            AccU acc;
            float bias = b3s[nt * 16 + ln];
            #pragma unroll
            for (int i = 0; i < 8; ++i) acc.x[i] = bias;
            #pragma unroll
            for (int kc = 0; kc < KC3; ++kc) {
                FragU a, bf;
                const uint4* ap =
                    (const uint4*)(h2 + ln * H2_STRIDE + kc * 32 + hi * 8);
                const uint4* ap2 =
                    (const uint4*)(h2 + ln * H2_STRIDE + kc * 32 + hi * 8 + 16);
                a.q[0] = ap[0]; a.q[1] = ap2[0];
                const uint4* bp =
                    (const uint4*)(w3s + ((size_t)(nt * KC3 + kc) * 32 + lane) * 8);
                bf.q[0] = bp[0]; bf.q[1] = bp[1];
                acc.f = __builtin_amdgcn_wmma_f32_16x16x32_f16(
                    false, a.h, false, bf.h, (short)0, acc.f, false, false);
            }
            int n = nt * 16 + ln;
            #pragma unroll
            for (int i = 0; i < 8; ++i) {
                int rr = rowBase + hi * 8 + i;
                if (rr < M_TOT) {
                    float s = 1.0f / (1.0f + __expf(-acc.x[i]));
                    int ob = rr / LPOS;
                    int op = rr - ob * LPOS;
                    out[((size_t)ob * N3 + n) * LPOS + op] = s;
                }
            }
        }
    }
}

extern "C" void kernel_launch(void* const* d_in, const int* in_sizes, int n_in,
                              void* d_out, int out_size, void* d_ws, size_t ws_size,
                              hipStream_t stream) {
    (void)in_sizes; (void)n_in; (void)out_size; (void)d_ws; (void)ws_size;
    const float* x  = (const float*)d_in[0];
    const float* w1 = (const float*)d_in[1];
    const float* b1 = (const float*)d_in[2];
    const float* w2 = (const float*)d_in[3];
    const float* b2 = (const float*)d_in[4];
    const float* w3 = (const float*)d_in[5];
    const float* b3 = (const float*)d_in[6];
    float* out = (float*)d_out;

    // CDNA5 allows 320 KB LDS per workgroup; opt in for >64 KB dynamic LDS.
    (void)hipFuncSetAttribute((const void*)mlpconv_fused_wmma,
                              hipFuncAttributeMaxDynamicSharedMemorySize,
                              SMEM_BYTES);
    dim3 grid(512), block(256);
    mlpconv_fused_wmma<<<grid, block, SMEM_BYTES, stream>>>(
        x, w1, b1, w2, b2, w3, b3, out);
}
